// CrossAttentionMasked_31791347925053
// MI455X (gfx1250) — matrix operand order repaired and checked by reading, not compiled, not run on hardware
//
#include <hip/hip_runtime.h>
#include <hip/hip_bf16.h>

// ---------------------------------------------------------------------------
// CrossAttentionMasked on gfx1250 (MI455X): all GEMM-like stages through
// v_wmma_f32_16x16x32_f16 (wave32), f32 accumulate.
//   - projection GEMMs: 128x128 block tile, 8 waves, wave tile 32x64
//     (8 WMMAs per 32-deep k-step), f32->f16 conversion during LDS staging
//   - flash attention: 8 waves / 128 query rows per block, K/V tiles shared
//     in LDS, C=80 padded to 96 in the WMMA K-dim, online softmax via
//     __shfl_xor within 16-lane half-groups (matches C/D fragment layout)
//   - masked query rows are simply zeroed post-attention (equivalent to the
//     reference's pre-softmax fill + post-attention zero; final GEMM then
//     yields exactly bout for those rows)
// Workspace: qh 20.97MB + kh 5.24MB + vh 5.24MB + oh 20.97MB + mask 64KB.
// ---------------------------------------------------------------------------

typedef _Float16 h8v  __attribute__((ext_vector_type(8)));
typedef _Float16 h16v __attribute__((ext_vector_type(16)));
typedef float    f8v  __attribute__((ext_vector_type(8)));

__device__ __forceinline__ h16v cat16(h8v a, h8v b) {
  return __builtin_shufflevector(a, b, 0,1,2,3,4,5,6,7,8,9,10,11,12,13,14,15);
}

__device__ __forceinline__ f8v wmma_f16(h16v a, h16v b, f8v c) {
  // D = A(16x32) * B(32x16) + C, f32 accumulate
  return __builtin_amdgcn_wmma_f32_16x16x32_f16(false, a, false, b,
                                                (short)0, c, false, false);
}

#define BM 128
#define BN 128
#define BK 32

// ---------------------------------------------------------------------------
// GEMM: out[M,Nc] = A[M,K] * B[K,Nc] (+bias). A is f32 or f16 (template),
// B is f32, output f16 or f32 (template). 256 threads = 8 waves; block tile
// 128x128; wave tile 32x64 -> 8 WMMAs per k-step.
// ---------------------------------------------------------------------------
template <typename AT, typename OT>
__global__ __launch_bounds__(256)
void gemm_kernel(const AT* __restrict__ A, const float* __restrict__ Bw,
                 const float* __restrict__ bias, OT* __restrict__ out,
                 int K, int Nc) {
  __shared__ _Float16 Asm[BM][BK];      // row-major tile of A
  __shared__ _Float16 Bsm[BN][BK];      // transposed tile: Bsm[n][k]

  const int t    = threadIdx.x;
  const int lane = t & 31;
  const int wave = t >> 5;
  const int g    = lane >> 4;           // half-group 0/1
  const int ml   = lane & 15;

  const int m0 = blockIdx.y * BM;
  const int n0 = blockIdx.x * BN;
  const int waveRow = (wave >> 1) * 32; // 0,32,64,96
  const int waveCol = (wave & 1) * 64;  // 0,64

  f8v acc[2][4];
#pragma unroll
  for (int i = 0; i < 2; ++i)
#pragma unroll
    for (int j = 0; j < 4; ++j) acc[i][j] = (f8v){};

  // staging indices: 4096 f16 per tile, 16 per thread
  const int ar = t >> 1;            // 0..127 (A row)
  const int ac = (t & 1) * 16;      // 0/16   (A col chunk)
  const int bk = t >> 3;            // 0..31  (B k row)
  const int bn = (t & 7) * 16;      // 0..112 (B n chunk)

  for (int k0 = 0; k0 < K; k0 += BK) {
    // ---- stage A (convert to f16), coalesced rows ----
    {
      const AT* ap = A + (size_t)(m0 + ar) * K + k0 + ac;
      h8v av0, av1;
#pragma unroll
      for (int i = 0; i < 8; ++i) { av0[i] = (_Float16)ap[i]; av1[i] = (_Float16)ap[8 + i]; }
      *(h8v*)&Asm[ar][ac]     = av0;
      *(h8v*)&Asm[ar][ac + 8] = av1;
      if (k0 + BK < K) __builtin_prefetch(ap + BK, 0, 1);
    }
    // ---- stage B transposed (convert to f16) ----
    {
      const float* bp = Bw + (size_t)(k0 + bk) * Nc + n0 + bn;
#pragma unroll
      for (int i = 0; i < 16; ++i) Bsm[bn + i][bk] = (_Float16)bp[i];
      if (k0 + BK < K) __builtin_prefetch(bp + (size_t)BK * Nc, 0, 1);
    }
    __syncthreads();

    // ---- fragments: A rows (K = (e<8?0:16)+8g+e%8), B cols (K = (e<8?0:16)+8g+e%8) ----
    h16v af[2];
#pragma unroll
    for (int mt = 0; mt < 2; ++mt) {
      h8v lo = *(const h8v*)&Asm[waveRow + mt * 16 + ml][g * 8];
      h8v hi = *(const h8v*)&Asm[waveRow + mt * 16 + ml][16 + g * 8];
      af[mt] = cat16(lo, hi);
    }
    h16v bf[4];
#pragma unroll
    for (int nt = 0; nt < 4; ++nt) {
      h8v lo = *(const h8v*)&Bsm[waveCol + nt * 16 + ml][g * 8];
      h8v hi = *(const h8v*)&Bsm[waveCol + nt * 16 + ml][16 + g * 8];
      bf[nt] = cat16(lo, hi);
    }
#pragma unroll
    for (int mt = 0; mt < 2; ++mt)
#pragma unroll
      for (int nt = 0; nt < 4; ++nt)
        acc[mt][nt] = wmma_f16(af[mt], bf[nt], acc[mt][nt]);
    __syncthreads();
  }

  // ---- store (C/D layout: row = r+8g, col = ml) ----
#pragma unroll
  for (int mt = 0; mt < 2; ++mt)
#pragma unroll
    for (int nt = 0; nt < 4; ++nt)
#pragma unroll
      for (int r = 0; r < 8; ++r) {
        const size_t row = (size_t)(m0 + waveRow + mt * 16 + r + 8 * g);
        const int col    = n0 + waveCol + nt * 16 + ml;
        float v = acc[mt][nt][r];
        if (bias) v += bias[col];
        out[row * Nc + col] = (OT)v;
      }
}

// ---------------------------------------------------------------------------
// Flash attention per (b,h): block = 8 waves = 128 query rows.
// ---------------------------------------------------------------------------
#define CDIM  80
#define CPAD  96
#define MT    32
#define NQ    4096
#define MK    1024
#define INNER 640

__global__ __launch_bounds__(256)
void attn_kernel(const _Float16* __restrict__ qh, const _Float16* __restrict__ kh,
                 const _Float16* __restrict__ vh, const float* __restrict__ maskf,
                 _Float16* __restrict__ oh) {
  __shared__ _Float16 Ksm[MT][CPAD];     // K tile rows, zero-padded cols 80..95
  __shared__ _Float16 Vsm[CDIM][MT];     // V tile transposed: Vsm[c][m]
  __shared__ _Float16 Psm[8][16][MT];    // per-wave P relayout buffer

  const int t    = threadIdx.x;
  const int lane = t & 31;
  const int wave = t >> 5;
  const int g    = lane >> 4;
  const int ml   = lane & 15;

  const int b  = blockIdx.z;
  const int h  = blockIdx.y;
  const int q0 = blockIdx.x * 128 + wave * 16;
  const float scale = rsqrtf((float)CDIM);   // C^-0.5

  // ---- Q fragments (3 K-steps of 32, last padded: c>=80 -> 0) ----
  const _Float16* qrow = qh + ((size_t)(b * NQ + q0 + ml) * INNER) + h * CDIM;
  h16v qa[3];
  {
    h8v z = {};
    h8v lo, hi;
    lo = *(const h8v*)(qrow + 0  + g * 8); hi = *(const h8v*)(qrow + 16 + g * 8);
    qa[0] = cat16(lo, hi);
    lo = *(const h8v*)(qrow + 32 + g * 8); hi = *(const h8v*)(qrow + 48 + g * 8);
    qa[1] = cat16(lo, hi);
    lo = *(const h8v*)(qrow + 64 + g * 8);                 // c = 64..79
    qa[2] = cat16(lo, z);                                  // c = 80..95 -> 0
  }

  f8v oacc[5];
#pragma unroll
  for (int i = 0; i < 5; ++i) oacc[i] = (f8v){};
  float mrun[8], lrun[8];
#pragma unroll
  for (int r = 0; r < 8; ++r) { mrun[r] = -3.0e38f; lrun[r] = 0.0f; }

  for (int m0 = 0; m0 < MK; m0 += MT) {
    __syncthreads();                  // protect LDS from previous iteration
    // stage K tile [32][96] with zero pad (uniform: 3072/256 = 12 iters)
    for (int idx = t; idx < MT * CPAD; idx += 256) {
      const int r  = idx / CPAD;
      const int cc = idx % CPAD;
      _Float16 val = (_Float16)0.0f;
      if (cc < CDIM)
        val = kh[((size_t)(b * MK + m0 + r) * INNER) + h * CDIM + cc];
      Ksm[r][cc] = val;
    }
    // stage V tile transposed [80][32] (uniform: 2560/256 = 10 iters)
    for (int idx = t; idx < CDIM * MT; idx += 256) {
      const int m = idx & (MT - 1);
      const int c = idx >> 5;
      Vsm[c][m] = vh[((size_t)(b * MK + m0 + m) * INNER) + h * CDIM + c];
    }
    if (m0 + MT < MK) {   // warm L2/WGP$ for the next tile (uniform branch)
      __builtin_prefetch(kh + ((size_t)(b * MK + m0 + MT + (t & 31)) * INNER) + h * CDIM, 0, 1);
      __builtin_prefetch(vh + ((size_t)(b * MK + m0 + MT + (t & 31)) * INNER) + h * CDIM, 0, 1);
    }
    __syncthreads();

    // ---- S = Q K^T for the two 16-column halves of this 32-wide M tile ----
    f8v s0 = {}, s1 = {};
#pragma unroll
    for (int s = 0; s < 3; ++s) {
      h8v kl = *(const h8v*)&Ksm[ml][s * 32 + g * 8];
      h8v kx = *(const h8v*)&Ksm[ml][s * 32 + 16 + g * 8];
      s0 = wmma_f16(qa[s], cat16(kl, kx), s0);
      kl = *(const h8v*)&Ksm[16 + ml][s * 32 + g * 8];
      kx = *(const h8v*)&Ksm[16 + ml][s * 32 + 16 + g * 8];
      s1 = wmma_f16(qa[s], cat16(kl, kx), s1);
    }

    // ---- online softmax (row r lives on the 16 lanes of half-group g) ----
    float corr[8];
#pragma unroll
    for (int r = 0; r < 8; ++r) {
      float v0 = s0[r] * scale;
      float v1 = s1[r] * scale;
      float mx = fmaxf(v0, v1);
#pragma unroll
      for (int off = 1; off < 16; off <<= 1)
        mx = fmaxf(mx, __shfl_xor(mx, off, 32));
      const float nm = fmaxf(mrun[r], mx);
      const float cr = __expf(mrun[r] - nm);
      const float p0 = __expf(v0 - nm);
      const float p1 = __expf(v1 - nm);
      float ps = p0 + p1;
#pragma unroll
      for (int off = 1; off < 16; off <<= 1)
        ps += __shfl_xor(ps, off, 32);
      lrun[r] = lrun[r] * cr + ps;
      mrun[r] = nm;
      corr[r] = cr;
      // write P (f16) to per-wave LDS for A-fragment relayout
      Psm[wave][r + 8 * g][ml]      = (_Float16)p0;
      Psm[wave][r + 8 * g][16 + ml] = (_Float16)p1;
    }
#pragma unroll
    for (int ct = 0; ct < 5; ++ct)
#pragma unroll
      for (int r = 0; r < 8; ++r) oacc[ct][r] *= corr[r];

    // ---- O += P(16x32) * V(32x80): P as A fragment, Vsm columns as B ----
    h8v pl = *(const h8v*)&Psm[wave][ml][g * 8];
    h8v ph = *(const h8v*)&Psm[wave][ml][16 + g * 8];
    h16v pf = cat16(pl, ph);
#pragma unroll
    for (int ct = 0; ct < 5; ++ct) {
      h8v vl = *(const h8v*)&Vsm[ct * 16 + ml][g * 8];
      h8v vx = *(const h8v*)&Vsm[ct * 16 + ml][16 + g * 8];
      oacc[ct] = wmma_f16(pf, cat16(vl, vx), oacc[ct]);
    }
  }

  // ---- normalize, apply query mask, store f16 for the output GEMM ----
#pragma unroll
  for (int r = 0; r < 8; ++r) {
    const int row = q0 + r + 8 * g;
    const float f = (1.0f / lrun[r]) * maskf[b * NQ + row];
    _Float16* op = oh + ((size_t)(b * NQ + row) * INNER) + h * CDIM;
#pragma unroll
    for (int ct = 0; ct < 5; ++ct)
      op[ct * 16 + ml] = (_Float16)(oacc[ct][r] * f);
  }
}

// ---------------------------------------------------------------------------
// qmask: box_mask is already 64x64 = N per batch (resize is identity).
// ---------------------------------------------------------------------------
__global__ void mask_kernel(const float* __restrict__ bm, float* __restrict__ mf,
                            int n) {
  const int i = blockIdx.x * 256 + threadIdx.x;
  if (i < n) mf[i] = (bm[i] > 0.5f) ? 1.0f : 0.0f;
}

// ---------------------------------------------------------------------------
extern "C" void kernel_launch(void* const* d_in, const int* in_sizes, int n_in,
                              void* d_out, int out_size, void* d_ws, size_t ws_size,
                              hipStream_t stream) {
  const float* x     = (const float*)d_in[0];   // [4,4096,640]
  const float* key   = (const float*)d_in[1];   // [4,1024,768]
  const float* value = (const float*)d_in[2];   // [4,1024,768]
  const float* box   = (const float*)d_in[3];   // [4,1,64,64]
  const float* Wq    = (const float*)d_in[4];   // [640,640]
  const float* Wk    = (const float*)d_in[5];   // [768,640]
  const float* Wv    = (const float*)d_in[6];   // [768,640]
  const float* Wout  = (const float*)d_in[7];   // [640,640]
  const float* bout  = (const float*)d_in[8];   // [640]
  float* out = (float*)d_out;                   // [4,4096,640]

  char* ws = (char*)d_ws;
  _Float16* qh = (_Float16*)ws; ws += (size_t)16384 * 640 * sizeof(_Float16);
  _Float16* kh = (_Float16*)ws; ws += (size_t)4096  * 640 * sizeof(_Float16);
  _Float16* vh = (_Float16*)ws; ws += (size_t)4096  * 640 * sizeof(_Float16);
  _Float16* oh = (_Float16*)ws; ws += (size_t)16384 * 640 * sizeof(_Float16);
  float* maskf = (float*)ws;    ws += (size_t)16384 * sizeof(float);

  mask_kernel<<<16384 / 256, 256, 0, stream>>>(box, maskf, 16384);

  // projections (f32 -> f16 compute -> f16 out); N=640 -> 5 col blocks of 128
  gemm_kernel<float, _Float16>
      <<<dim3(640 / BN, 16384 / BM), 256, 0, stream>>>(x,     Wq, nullptr, qh, 640, 640);
  gemm_kernel<float, _Float16>
      <<<dim3(640 / BN, 4096  / BM), 256, 0, stream>>>(key,   Wk, nullptr, kh, 768, 640);
  gemm_kernel<float, _Float16>
      <<<dim3(640 / BN, 4096  / BM), 256, 0, stream>>>(value, Wv, nullptr, vh, 768, 640);

  // attention: grid (N/128 query tiles, H, B), 8 waves per block
  attn_kernel<<<dim3(4096 / 128, 8, 4), 256, 0, stream>>>(qh, kh, vh, maskf, oh);

  // output projection + bias -> f32
  gemm_kernel<_Float16, float>
      <<<dim3(640 / BN, 16384 / BM), 256, 0, stream>>>(oh, Wout, bout, out, 640, 640);
}
